// DecoderRNN_Attn_12275016532026
// MI455X (gfx1250) — matrix operand order repaired and checked by reading, not compiled, not run on hardware
//
#include <hip/hip_runtime.h>

// ---------------------------------------------------------------------------
// DecoderRNN with Luong attention, 63 sequential steps.
// Strategy for MI455X (gfx1250):
//  - one-time fp32->bf16 conversion of all weights into workspace (~108 MB,
//    fits the 192MB L2); steps then stream weights from L2, not HBM
//  - all GEMMs via v_wmma_f32_16x16x32_bf16 (wave32, f32 accumulate);
//    each wave computes a 32x32 output tile (2 M-tiles x 2 N-tiles,
//    4 WMMAs per K-step) so A-fragment loads are amortized 2x
//  - bf16 activation buffers are written in-place into the next GEMM's
//    A-matrix slots by the pointwise kernels (no copy kernels)
// ---------------------------------------------------------------------------

typedef __attribute__((ext_vector_type(16))) __bf16 v16bf;
typedef __attribute__((ext_vector_type(8)))  __bf16 v8bf;
typedef __attribute__((ext_vector_type(8)))  float  v8f;

#define VSZ   32000
#define ESZ   512
#define HSZ   1024
#define BSZ   32
#define SSZ   64
#define TSZ   64
#define NSTEP (TSZ - 1)

// ------------------------- WMMA fragment loads -----------------------------
// 16-bit A 16x32 layout (ISA 7.12.2): lanes 0-15 row M=lane, K={0..7,16..23};
// lanes 16-31 row M=lane-16, K={8..15,24..31}.
__device__ __forceinline__ v16bf load_frag_a(const __bf16* A, int ldA,
                                             int m0, int k0, int lane) {
  int row  = m0 + (lane & 15);
  int half = lane >> 4;
  const __bf16* p = A + (size_t)row * ldA + k0 + half * 8;
  v8bf lo = *(const v8bf*)p;
  v8bf hi = *(const v8bf*)(p + 16);
  return __builtin_shufflevector(lo, hi, 0,1,2,3,4,5,6,7,8,9,10,11,12,13,14,15);
}

// B 32x16 (K-major per lane): lanes 0-15 col N=lane, K={0..7,16..23};
// lanes 16-31 col N=lane-16, K={8..15,24..31}.  B column n == row n of W[N,K].
__device__ __forceinline__ v16bf load_frag_b(const __bf16* W, int ldW,
                                             int n0, int k0, int lane) {
  int col  = n0 + (lane & 15);
  int half = lane >> 4;
  const __bf16* p = W + (size_t)col * ldW + k0 + half * 8;
  v8bf lo = *(const v8bf*)p;
  v8bf hi = *(const v8bf*)(p + 16);
  return __builtin_shufflevector(lo, hi, 0,1,2,3,4,5,6,7,8,9,10,11,12,13,14,15);
}

// ------------------------- generic M=32 WMMA GEMM --------------------------
// out[32,N] = A[32,K](bf16) * W[N,K](bf16)^T + bias.  128 thr = 4 waves.
// Each wave: 32 columns x 32 rows = 2 N-tiles x 2 M-tiles, 4 WMMAs per
// K-step of 32.  Block covers 128 columns; N % 128 == 0, K % 32 == 0.
__global__ __launch_bounds__(128) void gemm_bf16_kernel(
    const __bf16* __restrict__ A, int ldA,
    const __bf16* __restrict__ W, int ldW,
    const float* __restrict__ bias,
    float* __restrict__ out, int ldout, int K) {
  int lane = threadIdx.x & 31;
  int wave = threadIdx.x >> 5;
  int n0 = blockIdx.x * 128 + wave * 32;      // first N-tile
  int n1 = n0 + 16;                           // second N-tile
  v8f acc00 = {}, acc01 = {};                 // m0=0 : n0, n1
  v8f acc10 = {}, acc11 = {};                 // m0=16: n0, n1
  const __bf16* wrow0 = W + (size_t)(n0 + (lane & 15)) * ldW;
  const __bf16* wrow1 = W + (size_t)(n1 + (lane & 15)) * ldW;
  for (int k0 = 0; k0 < K; k0 += 32) {
    __builtin_prefetch(wrow0 + k0 + 256, 0, 3);  // near-cache prefetch on W
    __builtin_prefetch(wrow1 + k0 + 256, 0, 3);
    v16bf b0 = load_frag_b(W, ldW, n0, k0, lane);
    v16bf b1 = load_frag_b(W, ldW, n1, k0, lane);
    v16bf a0 = load_frag_a(A, ldA, 0, k0, lane);
    v16bf a1 = load_frag_a(A, ldA, 16, k0, lane);
    acc00 = __builtin_amdgcn_wmma_f32_16x16x32_bf16(false, a0, false, b0,
                                                    (short)0, acc00, false, false);
    acc01 = __builtin_amdgcn_wmma_f32_16x16x32_bf16(false, a0, false, b1,
                                                    (short)0, acc01, false, false);
    acc10 = __builtin_amdgcn_wmma_f32_16x16x32_bf16(false, a1, false, b0,
                                                    (short)0, acc10, false, false);
    acc11 = __builtin_amdgcn_wmma_f32_16x16x32_bf16(false, a1, false, b1,
                                                    (short)0, acc11, false, false);
  }
  int c0 = n0 + (lane & 15);
  int c1 = n1 + (lane & 15);
  float bv0 = bias ? bias[c0] : 0.0f;
  float bv1 = bias ? bias[c1] : 0.0f;
  int mbase = (lane >> 4) * 8;   // D layout: lanes16-31 hold M=8..15 / 24..31
#pragma unroll
  for (int v = 0; v < 8; ++v) {
    out[(size_t)(mbase + v) * ldout + c0]      = acc00[v] + bv0;
    out[(size_t)(mbase + v) * ldout + c1]      = acc01[v] + bv1;
    out[(size_t)(16 + mbase + v) * ldout + c0] = acc10[v] + bv0;
    out[(size_t)(16 + mbase + v) * ldout + c1] = acc11[v] + bv1;
  }
}

// ------------------------- setup kernels -----------------------------------
__global__ void cvt_f32_bf16_kernel(const float* __restrict__ s,
                                    __bf16* __restrict__ d, int n) {
  for (int i = blockIdx.x * blockDim.x + threadIdx.x; i < n;
       i += gridDim.x * blockDim.x)
    d[i] = (__bf16)s[i];
}

// concat [W_ih | W_hh] row-wise into one bf16 [4096, Kih+Khh]
__global__ void build_wcat_kernel(const float* __restrict__ Wih, int Kih,
                                  const float* __restrict__ Whh, int Khh,
                                  __bf16* __restrict__ dst) {
  int Kt = Kih + Khh;
  int total = 4 * HSZ * Kt;
  for (int i = blockIdx.x * blockDim.x + threadIdx.x; i < total;
       i += gridDim.x * blockDim.x) {
    int r = i / Kt, k = i - r * Kt;
    float v = (k < Kih) ? Wih[(size_t)r * Kih + k]
                        : Whh[(size_t)r * Khh + (k - Kih)];
    dst[i] = (__bf16)v;
  }
}

__global__ void bias_sum_kernel(const float* a, const float* b, float* o, int n) {
  int i = blockIdx.x * blockDim.x + threadIdx.x;
  if (i < n) o[i] = a[i] + b[i];
}

// init: attn_hidden=0 -> A0[:,512:1536]; h0[0]->A0[:,1536:2560];
//       h0[1]->A1[:,1024:2048]; c_state = c0
__global__ void init_state_kernel(const float* __restrict__ h0,
                                  const float* __restrict__ c0,
                                  __bf16* __restrict__ A0,
                                  __bf16* __restrict__ A1,
                                  float* __restrict__ cst) {
  int i = blockIdx.x * blockDim.x + threadIdx.x;  // 32*1024
  if (i >= BSZ * HSZ) return;
  int b = i >> 10, n = i & (HSZ - 1);
  A0[(size_t)b * 2560 + 512 + n]  = (__bf16)0.0f;
  A0[(size_t)b * 2560 + 1536 + n] = (__bf16)h0[i];                 // layer 0
  A1[(size_t)b * 2048 + 1024 + n] = (__bf16)h0[BSZ * HSZ + i];     // layer 1
  cst[i]             = c0[i];
  cst[BSZ * HSZ + i] = c0[BSZ * HSZ + i];
}

// ------------------------- per-step kernels --------------------------------
__global__ void embed_kernel(const int* __restrict__ tgt, int t,
                             const float* __restrict__ emb,
                             __bf16* __restrict__ A0) {
  int i = blockIdx.x * blockDim.x + threadIdx.x;  // 32*512
  if (i >= BSZ * ESZ) return;
  int b = i >> 9, e = i & (ESZ - 1);
  int w = tgt[b * TSZ + t];
  A0[(size_t)b * 2560 + e] = (__bf16)emb[(size_t)w * ESZ + e];
}

__device__ __forceinline__ float sigf(float x) { return 1.0f / (1.0f + __expf(-x)); }

// gates [32,4096] (i,f,g,o) -> c update; write h (bf16) to two destinations
__global__ void lstm_cell_kernel(const float* __restrict__ gates,
                                 float* __restrict__ c,
                                 __bf16* __restrict__ hA, int ldhA,
                                 __bf16* __restrict__ hB, int ldhB) {
  int i = blockIdx.x * blockDim.x + threadIdx.x;  // 32*1024
  if (i >= BSZ * HSZ) return;
  int b = i >> 10, n = i & (HSZ - 1);
  const float* g = gates + (size_t)b * 4 * HSZ;
  float gi = sigf(g[n]);
  float gf = sigf(g[n + HSZ]);
  float gg = tanhf(g[n + 2 * HSZ]);
  float go = sigf(g[n + 3 * HSZ]);
  float cn = gf * c[i] + gi * gg;
  c[i] = cn;
  float h = go * tanhf(cn);
  hA[(size_t)b * ldhA + n] = (__bf16)h;
  if (hB) hB[(size_t)b * ldhB + n] = (__bf16)h;
}

// scores -> softmax -> context, one block per batch row
__global__ __launch_bounds__(256) void attn_kernel(
    const float* __restrict__ enc,   // [S,B,H]
    const float* __restrict__ y,     // [B,H] = h1 @ Wa^T
    __bf16* __restrict__ ctx, int ldctx) {
  __shared__ float part[SSZ][4];
  __shared__ float al[SSZ];
  __shared__ float inv;
  int b = blockIdx.x, tid = threadIdx.x;
  int s = tid >> 2, q = tid & 3;
  const float* yb = y + (size_t)b * HSZ;
  const float* es = enc + ((size_t)s * BSZ + b) * HSZ;
  float p = 0.0f;
  for (int h = q; h < HSZ; h += 4) p += es[h] * yb[h];
  part[s][q] = p;
  __syncthreads();
  if (tid < SSZ) al[tid] = part[tid][0] + part[tid][1] + part[tid][2] + part[tid][3];
  __syncthreads();
  if (tid == 0) {
    float m = al[0];
    for (int i = 1; i < SSZ; ++i) m = fmaxf(m, al[i]);
    float sum = 0.0f;
    for (int i = 0; i < SSZ; ++i) { al[i] = __expf(al[i] - m); sum += al[i]; }
    inv = 1.0f / sum;
  }
  __syncthreads();
  float iv = inv;
  for (int h = tid; h < HSZ; h += 256) {
    float cacc = 0.0f;
    for (int s2 = 0; s2 < SSZ; ++s2)
      cacc += al[s2] * enc[((size_t)s2 * BSZ + b) * HSZ + h];
    ctx[(size_t)b * ldctx + h] = (__bf16)(cacc * iv);
  }
}

__global__ void tanh_bf16_kernel(const float* __restrict__ x,
                                 __bf16* __restrict__ d, int ldd) {
  int i = blockIdx.x * blockDim.x + threadIdx.x;  // 32*1024
  if (i >= BSZ * HSZ) return;
  int b = i >> 10, n = i & (HSZ - 1);
  d[(size_t)b * ldd + n] = (__bf16)tanhf(x[i]);
}

// log-softmax over V=32000 + argmax, one block per batch row
__global__ __launch_bounds__(256) void logsoftmax_argmax_kernel(
    const float* __restrict__ logits, float* __restrict__ out,
    long long* __restrict__ words) {
  __shared__ float smax[256];
  __shared__ int   sidx[256];
  __shared__ float ssum[256];
  int b = blockIdx.x, tid = threadIdx.x;
  const float* lb = logits + (size_t)b * VSZ;
  float* ob = out + (size_t)b * VSZ;
  float m = -3.402823e38f; int mi = 0;
  for (int v = tid; v < VSZ; v += 256) {
    float x = lb[v];
    if (x > m) { m = x; mi = v; }
  }
  smax[tid] = m; sidx[tid] = mi;
  __syncthreads();
  for (int off = 128; off > 0; off >>= 1) {
    if (tid < off) {
      float o = smax[tid + off];
      if (o > smax[tid] || (o == smax[tid] && sidx[tid + off] < sidx[tid])) {
        smax[tid] = o; sidx[tid] = sidx[tid + off];
      }
    }
    __syncthreads();
  }
  float gmax = smax[0];
  float s = 0.0f;
  for (int v = tid; v < VSZ; v += 256) s += __expf(lb[v] - gmax);
  ssum[tid] = s;
  __syncthreads();
  for (int off = 128; off > 0; off >>= 1) {
    if (tid < off) ssum[tid] += ssum[tid + off];
    __syncthreads();
  }
  float lse = gmax + __logf(ssum[0]);
  for (int v = tid; v < VSZ; v += 256) ob[v] = lb[v] - lse;
  if (tid == 0) words[b] = (long long)sidx[0];
}

// ---------------------------------------------------------------------------
extern "C" void kernel_launch(void* const* d_in, const int* in_sizes, int n_in,
                              void* d_out, int out_size, void* d_ws, size_t ws_size,
                              hipStream_t stream) {
  const int*   tgt    = (const int*)  d_in[0];
  const float* enc    = (const float*)d_in[1];
  const float* h0     = (const float*)d_in[2];
  const float* c0     = (const float*)d_in[3];
  const float* emb    = (const float*)d_in[4];
  const float* W_ih0  = (const float*)d_in[5];
  const float* W_hh0  = (const float*)d_in[6];
  const float* b_ih0  = (const float*)d_in[7];
  const float* b_hh0  = (const float*)d_in[8];
  const float* W_ih1  = (const float*)d_in[9];
  const float* W_hh1  = (const float*)d_in[10];
  const float* b_ih1  = (const float*)d_in[11];
  const float* b_hh1  = (const float*)d_in[12];
  const float* Wa     = (const float*)d_in[13];
  const float* Wcc    = (const float*)d_in[14];
  const float* b_cc   = (const float*)d_in[15];
  const float* Wout   = (const float*)d_in[16];
  const float* b_out  = (const float*)d_in[17];

  // ---- carve workspace (aligned to 256B) ----
  char* p = (char*)d_ws;
  auto carve = [&](size_t bytes) -> void* {
    void* r = (void*)p;
    p += (bytes + 255) & ~(size_t)255;
    return r;
  };
  __bf16* Wcat0   = (__bf16*)carve((size_t)4 * HSZ * 2560 * 2);   // [4096,2560]
  __bf16* Wcat1   = (__bf16*)carve((size_t)4 * HSZ * 2048 * 2);   // [4096,2048]
  __bf16* Wa_bf   = (__bf16*)carve((size_t)HSZ * HSZ * 2);        // [1024,1024]
  __bf16* Wcc_bf  = (__bf16*)carve((size_t)HSZ * 2 * HSZ * 2);    // [1024,2048]
  __bf16* Wout_bf = (__bf16*)carve((size_t)VSZ * HSZ * 2);        // [32000,1024]
  float*  bias0   = (float*) carve(4 * HSZ * 4);
  float*  bias1   = (float*) carve(4 * HSZ * 4);
  __bf16* A0      = (__bf16*)carve((size_t)BSZ * 2560 * 2);  // [emb|ah|h0prev]
  __bf16* A1      = (__bf16*)carve((size_t)BSZ * 2048 * 2);  // [h0new|h1prev]
  __bf16* C2      = (__bf16*)carve((size_t)BSZ * 2048 * 2);  // [h1new|context]
  float*  cst     = (float*) carve((size_t)2 * BSZ * HSZ * 4);
  float*  gates   = (float*) carve((size_t)BSZ * 4 * HSZ * 4);
  float*  ybuf    = (float*) carve((size_t)BSZ * HSZ * 4);
  float*  preah   = (float*) carve((size_t)BSZ * HSZ * 4);
  float*  logits  = (float*) carve((size_t)BSZ * VSZ * 4);

  float* outf = (float*)d_out;                               // [63,32,32000]
  long long* wordsp = (long long*)(outf + (size_t)NSTEP * BSZ * VSZ);  // [63,32]

  // ---- one-time weight prep (bf16 conversion; ~108MB -> fits L2) ----
  build_wcat_kernel<<<2048, 256, 0, stream>>>(W_ih0, ESZ + HSZ, W_hh0, HSZ, Wcat0);
  build_wcat_kernel<<<2048, 256, 0, stream>>>(W_ih1, HSZ, W_hh1, HSZ, Wcat1);
  cvt_f32_bf16_kernel<<<1024, 256, 0, stream>>>(Wa,   Wa_bf,  HSZ * HSZ);
  cvt_f32_bf16_kernel<<<1024, 256, 0, stream>>>(Wcc,  Wcc_bf, HSZ * 2 * HSZ);
  cvt_f32_bf16_kernel<<<4096, 256, 0, stream>>>(Wout, Wout_bf, VSZ * HSZ);
  bias_sum_kernel<<<16, 256, 0, stream>>>(b_ih0, b_hh0, bias0, 4 * HSZ);
  bias_sum_kernel<<<16, 256, 0, stream>>>(b_ih1, b_hh1, bias1, 4 * HSZ);
  init_state_kernel<<<128, 256, 0, stream>>>(h0, c0, A0, A1, cst);

  // ---- 63 sequential decoder steps ----
  for (int t = 0; t < NSTEP; ++t) {
    embed_kernel<<<64, 256, 0, stream>>>(tgt, t, emb, A0);
    // LSTM layer 0: gates = [emb|ah|h0] @ [W_ih0|W_hh0]^T
    gemm_bf16_kernel<<<32, 128, 0, stream>>>(A0, 2560, Wcat0, 2560, bias0,
                                             gates, 4 * HSZ, 2560);
    lstm_cell_kernel<<<128, 256, 0, stream>>>(gates, cst, A1, 2048,
                                              A0 + 1536, 2560);
    // LSTM layer 1: gates = [h0new|h1prev] @ [W_ih1|W_hh1]^T
    gemm_bf16_kernel<<<32, 128, 0, stream>>>(A1, 2048, Wcat1, 2048, bias1,
                                             gates, 4 * HSZ, 2048);
    lstm_cell_kernel<<<128, 256, 0, stream>>>(gates, cst + BSZ * HSZ,
                                              A1 + 1024, 2048, C2, 2048);
    // y = h1 @ Wa^T   (A = C2[:, :1024], ld 2048)
    gemm_bf16_kernel<<<8, 128, 0, stream>>>(C2, 2048, Wa_bf, HSZ, nullptr,
                                            ybuf, HSZ, HSZ);
    attn_kernel<<<32, 256, 0, stream>>>(enc, ybuf, C2 + 1024, 2048);
    // ah = tanh([h1|ctx] @ W_concat^T + b)
    gemm_bf16_kernel<<<8, 128, 0, stream>>>(C2, 2048, Wcc_bf, 2 * HSZ, b_cc,
                                            preah, HSZ, 2 * HSZ);
    tanh_bf16_kernel<<<128, 256, 0, stream>>>(preah, A0 + 512, 2560);
    // logits = ah @ W_out^T + b_out   (A = A0[:, 512:1536], ld 2560)
    gemm_bf16_kernel<<<250, 128, 0, stream>>>(A0 + 512, 2560, Wout_bf, HSZ,
                                              b_out, logits, VSZ, HSZ);
    logsoftmax_argmax_kernel<<<32, 256, 0, stream>>>(
        logits, outf + (size_t)t * BSZ * VSZ, wordsp + (size_t)t * BSZ);
  }
}
